// CCNNCAModel_12996571037789
// MI455X (gfx1250) — matrix operand order repaired and checked by reading, not compiled, not run on hardware
//
#include <hip/hip_runtime.h>
#include <math.h>

// ---------------------------------------------------------------------------
// CDNA5 (gfx1250, wave32) implementation.
// k0: one-shot build of pre-swizzled bf16 hi/lo W-fragment image (64 KB) in ws
// k1: column sums of Q = (1/16)[cos(ZW), sin(ZW)]   (bf16 hi/lo split WMMA)
// k2: alpha softmax, Atil = (alpha/16)*A, reg_loss (3x3 eigen closed form)
// k3: recompute Q tiles, predictions = Qtilde @ A, AW = row softmax
// W image and Z tiles are streamed HBM->LDS with global_load_async_to_lds_b128.
// ---------------------------------------------------------------------------

typedef __attribute__((ext_vector_type(16))) __bf16 v16bf;
typedef __attribute__((ext_vector_type(8)))  float  v8f;
typedef __attribute__((ext_vector_type(4)))  float  f4;

#define RFF   256
#define DIN   64
#define NBLK  512
#define WAVES 8
#define TPB   256
#define WFRAG_FLOATS (16 * 2 * 2 * 32 * 8)   // 64 KB as floats (16384)
#define USE_ASYNC 1

// v_sin_f32 / v_cos_f32 take input in revolutions (x / 2pi)
__device__ __forceinline__ float fcos(float x) {
  return __builtin_amdgcn_cosf(x * 0.15915494309189535f);
}
__device__ __forceinline__ float fsin(float x) {
  return __builtin_amdgcn_sinf(x * 0.15915494309189535f);
}

__device__ __forceinline__ v8f wmma_bf16(v16bf a, v16bf b, v8f c) {
  // (neg_a, A, neg_b, B, c_mod, C, reuse_a, reuse_b)
  return __builtin_amdgcn_wmma_f32_16x16x32_bf16(false, a, false, b,
                                                 (short)0, c, false, false);
}

// ---- async HBM -> LDS: one 16x64 fp32 Z tile (4 KB), ASYNCcnt += 8 ---------
__device__ __forceinline__ void async_tile(const float* gsrc, float* ldst, int lane) {
#if USE_ASYNC
  unsigned long long ga = (unsigned long long)(uintptr_t)gsrc + (unsigned)(lane * 16);
  unsigned la = (unsigned)(uintptr_t)ldst + (unsigned)(lane * 16);
#pragma unroll
  for (int c = 0; c < 8; ++c) {
    asm volatile("global_load_async_to_lds_b128 %0, %1, off"
                 :: "v"(la + (unsigned)(c * 512)),
                    "v"(ga + (unsigned long long)(c * 512))
                 : "memory");
  }
#else
  f4* dst = (f4*)ldst;
  const f4* src = (const f4*)gsrc;
#pragma unroll
  for (int c = 0; c < 8; ++c) dst[c * 32 + lane] = src[c * 32 + lane];
#endif
}

// ---- async HBM -> LDS: 64 KB W-fragment image, ASYNCcnt += 16 per wave -----
__device__ __forceinline__ void async_wimage(const float* gsrc, void* ldst, int tid) {
#if USE_ASYNC
  unsigned long long ga = (unsigned long long)(uintptr_t)gsrc + (unsigned)(tid * 16);
  unsigned la = (unsigned)(uintptr_t)ldst + (unsigned)(tid * 16);
#pragma unroll
  for (int k = 0; k < 16; ++k) {
    asm volatile("global_load_async_to_lds_b128 %0, %1, off"
                 :: "v"(la + (unsigned)(k * 4096)),
                    "v"(ga + (unsigned long long)(k * 4096))
                 : "memory");
  }
#else
  f4* dst = (f4*)ldst;
  const f4* src = (const f4*)gsrc;
#pragma unroll
  for (int k = 0; k < 16; ++k) dst[k * 256 + tid] = src[k * 256 + tid];
#endif
}

__device__ __forceinline__ void wait_async(bool pipelined) {
#if USE_ASYNC
  if (pipelined) asm volatile("s_wait_asynccnt 0x8" ::: "memory");  // newest 8 may remain
  else           asm volatile("s_wait_asynccnt 0x0" ::: "memory");  // drain
#endif
}

// A 16x32 bf16 layout (wave32): lanes 0-15 and 16-31 both cover rows 0..15
// (row = lane & 15); lane-high half takes the K+8 / K+24 subranges.
// zrow points into the LDS-staged tile -> ds_load_b128 reads.
__device__ __forceinline__ void load_afrag(const float* zrow, int laneHi,
                                           v16bf* ahi, v16bf* alo) {
#pragma unroll
  for (int s = 0; s < 2; ++s) {
    const int kb0 = 32 * s + laneHi * 8;
    f4 a = *(const f4*)(zrow + kb0);
    f4 b = *(const f4*)(zrow + kb0 + 4);
    f4 c = *(const f4*)(zrow + kb0 + 16);
    f4 d = *(const f4*)(zrow + kb0 + 20);
    float f[16] = {a[0], a[1], a[2], a[3], b[0], b[1], b[2], b[3],
                   c[0], c[1], c[2], c[3], d[0], d[1], d[2], d[3]};
#pragma unroll
    for (int j = 0; j < 16; ++j) {
      __bf16 h  = (__bf16)f[j];
      ahi[s][j] = h;
      alo[s][j] = (__bf16)(f[j] - (float)h);
    }
  }
}

// ---------------------------------------------------------------------------
// Kernel 0 (one-shot): W (64x256 fp32) -> pre-swizzled bf16 hi/lo B-fragment
// image in ws. B 32x16 bf16 layout (wave32): lanes 0-15 hold K=0..7,16..23, lanes
// 16-31 hold K=8..15,24..31; column = lane & 15; 16 bf16 per lane = one v16bf.
// Image layout: [ctile 16][kstep 2][hi/lo 2][lane 32] of v16bf (32 B).
// ---------------------------------------------------------------------------
__global__ __launch_bounds__(TPB) void k_wfrag(const float* __restrict__ W,
                                               float* __restrict__ wfrag) {
  __bf16* wf = (__bf16*)wfrag;
  const int start = blockIdx.x * TPB + threadIdx.x;
  for (int e = start; e < 16 * 2 * 32 * 16; e += gridDim.x * TPB) {
    int c    = e >> 10;
    int rem  = e & 1023;
    int s    = rem >> 9;  rem &= 511;
    int lane = rem >> 4;
    int j    = rem & 15;
    int K    = 32 * s + ((lane >= 16) ? 8 : 0) + j + ((j >= 8) ? 8 : 0);
    int col  = c * 16 + (lane & 15);
    float w  = W[(size_t)K * RFF + col];
    __bf16 h = (__bf16)w;
    wf[((((c * 2 + s) * 2 + 0) * 32 + lane) << 4) + j] = h;
    wf[((((c * 2 + s) * 2 + 1) * 32 + lane) << 4) + j] = (__bf16)(w - (float)h);
  }
}

// ---------------------------------------------------------------------------
// Kernel 1: column sums of cos/sin(ZW) -> per-block partials [NBLK][512]
// ---------------------------------------------------------------------------
__global__ __launch_bounds__(TPB, 1) void k_rff_colsum(const float* __restrict__ Z,
                                                       const float* __restrict__ wfrag,
                                                       float* __restrict__ partials,
                                                       int N) {
  __shared__ v16bf s_wf[16][2][2][32];       // 64 KB: [ctile][kstep][hi/lo][lane]
  __shared__ float s_z[WAVES][2][16 * DIN];  // 64 KB: double-buffered Z tiles
  __shared__ float s_part[WAVES][2 * RFF];   // 16 KB
  const int tid = threadIdx.x;
  const int lane = tid & 31, wave = tid >> 5;
  const int laneHi = lane >> 4, lcol = lane & 15;

  const int base0  = (blockIdx.x * WAVES + wave) * 16;
  const int stride = gridDim.x * WAVES * 16;

  // Stage W-fragment image (16 asyncs/wave) then first Z tile (8 asyncs);
  // in-order completion => wait<=8 drains exactly the W chunks.
  async_wimage(wfrag, &s_wf[0][0][0][0], tid);
  const bool t0 = (base0 + 16 <= N);
  if (t0) async_tile(Z + (size_t)base0 * DIN, &s_z[wave][0][0], lane);
  wait_async(t0);
  __syncthreads();

  float csum[16], ssum[16];
#pragma unroll
  for (int c = 0; c < 16; ++c) { csum[c] = 0.f; ssum[c] = 0.f; }

  int buf = 0;
  for (int base = base0; base + 16 <= N; base += stride, buf ^= 1) {
    const int nbase = base + stride;
    const bool have_next = (nbase + 16 <= N);
    if (have_next) async_tile(Z + (size_t)nbase * DIN, &s_z[wave][buf ^ 1][0], lane);
    wait_async(have_next);

    const float* zrow = &s_z[wave][buf][lcol * DIN];
    v16bf ahi[2], alo[2];
    load_afrag(zrow, laneHi, ahi, alo);

#pragma unroll 2
    for (int c = 0; c < 16; ++c) {
      v8f acc = {0.f, 0.f, 0.f, 0.f, 0.f, 0.f, 0.f, 0.f};
#pragma unroll
      for (int s = 0; s < 2; ++s) {
        v16bf bh = s_wf[c][s][0][lane];
        v16bf bl = s_wf[c][s][1][lane];
        acc = wmma_bf16(ahi[s], bh, acc);  // hi*hi
        acc = wmma_bf16(ahi[s], bl, acc);  // hi*lo
        acc = wmma_bf16(alo[s], bh, acc);  // lo*hi  (~fp32 accuracy)
      }
      float pc = 0.f, ps = 0.f;
#pragma unroll
      for (int r = 0; r < 8; ++r) {  // rows laneHi*8 + r of the tile
        float p = acc[r];
        pc += fcos(p);
        ps += fsin(p);
      }
      csum[c] += pc;
      ssum[c] += ps;
    }
  }

  // lane L and L^16 cover the same column (different row halves)
#pragma unroll
  for (int c = 0; c < 16; ++c) {
    csum[c] += __shfl_xor(csum[c], 16);
    ssum[c] += __shfl_xor(ssum[c], 16);
  }
  if (laneHi == 0) {
#pragma unroll
    for (int c = 0; c < 16; ++c) {
      s_part[wave][c * 16 + lcol]       = csum[c];
      s_part[wave][RFF + c * 16 + lcol] = ssum[c];
    }
  }
  __syncthreads();
  for (int f = tid; f < 2 * RFF; f += TPB) {
    float t = 0.f;
#pragma unroll
    for (int w = 0; w < WAVES; ++w) t += s_part[w][f];
    partials[(size_t)blockIdx.x * (2 * RFF) + f] = t;  // deterministic
  }
}

// ---------------------------------------------------------------------------
// Kernel 2 (1 block): alpha softmax, Atil = (alpha/16)*A, reg_loss (nuclear)
// ---------------------------------------------------------------------------
__global__ __launch_bounds__(TPB) void k_attn(const float* __restrict__ partials,
                                              const float* __restrict__ A,
                                              float* __restrict__ atil,
                                              float* __restrict__ out,
                                              int N, int nblk) {
  __shared__ float cs[2 * RFF];
  __shared__ float red[TPB];
  __shared__ float s_alpha[RFF];
  __shared__ float g[9];
  const int tid = threadIdx.x;

  for (int f = tid; f < 2 * RFF; f += TPB) {
    float t = 0.f;
    for (int b = 0; b < nblk; ++b) t += partials[(size_t)b * (2 * RFF) + f];
    cs[f] = t * (0.0625f / (float)N);  // column mean of Q (sqrt(1/R) folded in)
  }
  __syncthreads();

  const float Am0 = (A[0] + A[1] + A[2]) * (1.f / 3.f);  // A_mean[0]
  const float Am1 = (A[3] + A[4] + A[5]) * (1.f / 3.f);  // A_mean[1]
  // scores[k]/(0.1*sqrt(256)) = scores[k]*0.625
  float logit = (Am0 * cs[2 * tid] + Am1 * cs[2 * tid + 1]) * 0.625f;

  red[tid] = logit;
  __syncthreads();
  for (int s = TPB / 2; s > 0; s >>= 1) {
    if (tid < s) red[tid] = fmaxf(red[tid], red[tid + s]);
    __syncthreads();
  }
  float m = red[0];
  __syncthreads();
  float e = expf(logit - m);
  red[tid] = e;
  __syncthreads();
  for (int s = TPB / 2; s > 0; s >>= 1) {
    if (tid < s) red[tid] += red[tid + s];
    __syncthreads();
  }
  float alpha = e / red[0];
  s_alpha[tid] = alpha;
  out[(size_t)6 * N + 1 + tid] = alpha;  // alpha output [256]
  __syncthreads();

  // Atil[f][o] = alpha[f>>1] * (1/16) * A[f][o]   (Q scale folded here)
  for (int f = tid; f < 2 * RFF; f += TPB) {
    float al = s_alpha[f >> 1] * 0.0625f;
    atil[f * 3 + 0] = al * A[f * 3 + 0];
    atil[f * 3 + 1] = al * A[f * 3 + 1];
    atil[f * 3 + 2] = al * A[f * 3 + 2];
  }

  // Gram matrix G = A^T A (3x3) for the nuclear norm
  if (tid < 9) {
    int i = tid / 3, j = tid % 3;
    float t = 0.f;
    for (int f = 0; f < 2 * RFF; ++f) t += A[f * 3 + i] * A[f * 3 + j];
    g[tid] = t;
  }
  __syncthreads();
  if (tid == 0) {
    float a00 = g[0], a01 = g[1], a02 = g[2], a11 = g[4], a12 = g[5], a22 = g[8];
    float q  = (a00 + a11 + a22) * (1.f / 3.f);
    float p1 = a01 * a01 + a02 * a02 + a12 * a12;
    float p2 = (a00 - q) * (a00 - q) + (a11 - q) * (a11 - q) +
               (a22 - q) * (a22 - q) + 2.f * p1;
    float nuc;
    if (p2 <= 1e-30f) {
      nuc = sqrtf(fmaxf(a00, 0.f)) + sqrtf(fmaxf(a11, 0.f)) + sqrtf(fmaxf(a22, 0.f));
    } else {
      float p   = sqrtf(p2 * (1.f / 6.f));
      float ip  = 1.f / p;
      float b00 = (a00 - q) * ip, b11 = (a11 - q) * ip, b22 = (a22 - q) * ip;
      float b01 = a01 * ip, b02 = a02 * ip, b12 = a12 * ip;
      float detB = b00 * (b11 * b22 - b12 * b12) - b01 * (b01 * b22 - b12 * b02) +
                   b02 * (b01 * b12 - b11 * b02);
      float r   = fminf(fmaxf(detB * 0.5f, -1.f), 1.f);
      float phi = acosf(r) * (1.f / 3.f);
      float e1  = q + 2.f * p * cosf(phi);
      float e3  = q + 2.f * p * cosf(phi + 2.0943951023931953f);
      float e2  = 3.f * q - e1 - e3;
      nuc = sqrtf(fmaxf(e1, 0.f)) + sqrtf(fmaxf(e2, 0.f)) + sqrtf(fmaxf(e3, 0.f));
    }
    out[(size_t)6 * N] = 0.01f * nuc;  // reg_loss
  }
}

// ---------------------------------------------------------------------------
// Kernel 3: recompute Q tiles, predictions = Qtilde@A, AW = softmax(pred/16)
// ---------------------------------------------------------------------------
__global__ __launch_bounds__(TPB, 1) void k_predict(const float* __restrict__ Z,
                                                    const float* __restrict__ wfrag,
                                                    const float* __restrict__ atil,
                                                    float* __restrict__ out, int N) {
  __shared__ v16bf s_wf[16][2][2][32];       // 64 KB
  __shared__ float s_z[WAVES][2][16 * DIN];  // 64 KB
  __shared__ float s_atil[2 * RFF * 3];      // 6 KB
  const int tid = threadIdx.x;
  const int lane = tid & 31, wave = tid >> 5;
  const int laneHi = lane >> 4, lcol = lane & 15;

  const int base0  = (blockIdx.x * WAVES + wave) * 16;
  const int stride = gridDim.x * WAVES * 16;

  async_wimage(wfrag, &s_wf[0][0][0][0], tid);
  const bool t0 = (base0 + 16 <= N);
  if (t0) async_tile(Z + (size_t)base0 * DIN, &s_z[wave][0][0], lane);
  for (int i = tid; i < 2 * RFF * 3; i += TPB) s_atil[i] = atil[i];
  wait_async(t0);
  __syncthreads();

  float* outP  = out;
  float* outAW = out + (size_t)3 * N;

  int buf = 0;
  for (int base = base0; base + 16 <= N; base += stride, buf ^= 1) {
    const int nbase = base + stride;
    const bool have_next = (nbase + 16 <= N);
    if (have_next) async_tile(Z + (size_t)nbase * DIN, &s_z[wave][buf ^ 1][0], lane);
    wait_async(have_next);

    const float* zrow = &s_z[wave][buf][lcol * DIN];
    v16bf ahi[2], alo[2];
    load_afrag(zrow, laneHi, ahi, alo);

    float pr[8][3];
#pragma unroll
    for (int r = 0; r < 8; ++r) { pr[r][0] = 0.f; pr[r][1] = 0.f; pr[r][2] = 0.f; }

#pragma unroll 2
    for (int c = 0; c < 16; ++c) {
      v8f acc = {0.f, 0.f, 0.f, 0.f, 0.f, 0.f, 0.f, 0.f};
#pragma unroll
      for (int s = 0; s < 2; ++s) {
        v16bf bh = s_wf[c][s][0][lane];
        v16bf bl = s_wf[c][s][1][lane];
        acc = wmma_bf16(ahi[s], bh, acc);
        acc = wmma_bf16(ahi[s], bl, acc);
        acc = wmma_bf16(alo[s], bh, acc);
      }
      const int col = c * 16 + lcol;
      float ac0 = s_atil[col * 3 + 0], ac1 = s_atil[col * 3 + 1], ac2 = s_atil[col * 3 + 2];
      float as0 = s_atil[(RFF + col) * 3 + 0], as1 = s_atil[(RFF + col) * 3 + 1],
            as2 = s_atil[(RFF + col) * 3 + 2];
#pragma unroll
      for (int r = 0; r < 8; ++r) {
        float p  = acc[r];
        float qc = fcos(p), qs = fsin(p);  // 1/16 scale folded into atil
        pr[r][0] += qc * ac0 + qs * as0;
        pr[r][1] += qc * ac1 + qs * as1;
        pr[r][2] += qc * ac2 + qs * as2;
      }
    }

    // butterfly-reduce the 16 column-lanes within each half of the wave
#pragma unroll
    for (int off = 1; off < 16; off <<= 1) {
#pragma unroll
      for (int r = 0; r < 8; ++r) {
        pr[r][0] += __shfl_xor(pr[r][0], off);
        pr[r][1] += __shfl_xor(pr[r][1], off);
        pr[r][2] += __shfl_xor(pr[r][2], off);
      }
    }

    if (lcol < 8) {
      const int r   = lcol;
      const int row = base + laneHi * 8 + r;
      float p0 = pr[r][0], p1 = pr[r][1], p2 = pr[r][2];
      outP[(size_t)row * 3 + 0] = p0;
      outP[(size_t)row * 3 + 1] = p1;
      outP[(size_t)row * 3 + 2] = p2;
      float m  = fmaxf(p0, fmaxf(p1, p2));
      float e0 = expf((p0 - m) * 0.0625f);
      float e1 = expf((p1 - m) * 0.0625f);
      float e2 = expf((p2 - m) * 0.0625f);
      float inv = 1.f / (e0 + e1 + e2);
      outAW[(size_t)row * 3 + 0] = e0 * inv;
      outAW[(size_t)row * 3 + 1] = e1 * inv;
      outAW[(size_t)row * 3 + 2] = e2 * inv;
    }
  }
}

// ---------------------------------------------------------------------------
extern "C" void kernel_launch(void* const* d_in, const int* in_sizes, int n_in,
                              void* d_out, int out_size, void* d_ws, size_t ws_size,
                              hipStream_t stream) {
  const float* Z = (const float*)d_in[0];  // [N, 64]
  const float* W = (const float*)d_in[1];  // [64, 256]
  const float* A = (const float*)d_in[2];  // [512, 3]
  float* out = (float*)d_out;              // pred[3N] | AW[3N] | reg[1] | alpha[256]
  float* ws  = (float*)d_ws;

  const int N = in_sizes[0] / DIN;
  float* partials = ws;                              // NBLK * 512 floats (1 MB)
  float* atil     = ws + (size_t)NBLK * 2 * RFF;     // 1536 floats
  float* wfrag    = atil + 2 * RFF * 3;              // 16384 floats (64 KB image)

  k_wfrag<<<16, TPB, 0, stream>>>(W, wfrag);
  k_rff_colsum<<<NBLK, TPB, 0, stream>>>(Z, wfrag, partials, N);
  k_attn<<<1, TPB, 0, stream>>>(partials, A, atil, out, N, NBLK);
  k_predict<<<NBLK, TPB, 0, stream>>>(Z, wfrag, atil, out, N);
}
